// GINEEdgeLayer_21801253994646
// MI455X (gfx1250) — compile-verified, matrix operand order checked by reading, not run on hardware
//
#include <hip/hip_runtime.h>
#include <hip/hip_bf16.h>

typedef __attribute__((ext_vector_type(16))) __bf16          v16bf;
typedef __attribute__((ext_vector_type(8)))  float           v8f;

#define E_EDGES   160000
#define D_DIM     256
#define K_TOT     768
#define NGRAPH    64
#define GN_EPS    1e-5f

// ---- workspace layout (bytes) ----
#define WS_WB     0u        // bf16[24*16*32*16] = 196608 elems = 393216 B (W1 in WMMA-B fragment order)
#define WS_SEG    393216u   // int[160000] = 640000 B
#define WS_CNT    1033216u  // float[64]
#define WS_SUM    1033472u  // float[64*256]
#define WS_SQ     1099008u  // float[64*256]
#define WS_SCALE  1164544u  // float[64*256]
#define WS_SHIFT  1230080u  // float[64*256]  (end 1295616)

__device__ __forceinline__ unsigned short f2bf(float f) {
    unsigned u = __float_as_uint(f);
    unsigned r = u + 0x7FFFu + ((u >> 16) & 1u);   // round-to-nearest-even
    return (unsigned short)(r >> 16);
}

// ---------------------------------------------------------------------------
// Kernel 1: W1 (f32 [768][256], row-major) -> bf16 in WMMA B-fragment order.
// Output flat index o = ((kc*16 + nsub)*32 + lane)*16 + j
//   K = kc*32 + (j&7) + 8*(lane>>4) + 16*(j>>3),  N = nsub*16 + (lane&15)
// ---------------------------------------------------------------------------
__global__ void k_w1_to_frag(const float* __restrict__ W1,
                             unsigned short* __restrict__ wBf) {
    int o    = blockIdx.x * 256 + threadIdx.x;          // 196608 total, exact
    int j    = o & 15;
    int lane = (o >> 4) & 31;
    int nsub = (o >> 9) & 15;
    int kc   = o >> 13;
    int K = kc * 32 + (j & 7) + ((lane >> 4) << 3) + ((j >> 3) << 4);
    int n = nsub * 16 + (lane & 15);
    wBf[o] = f2bf(W1[K * D_DIM + n]);
}

// ---------------------------------------------------------------------------
// Kernel 2: per-edge segment id + per-graph edge count
// ---------------------------------------------------------------------------
__global__ void k_seg(const int* __restrict__ ei, const int* __restrict__ batch,
                      int* __restrict__ seg, float* __restrict__ cnt) {
    int t = blockIdx.x * 256 + threadIdx.x;
    if (t < E_EDGES) {
        int s = batch[ei[t]];        // batch[src[t]]
        seg[t] = s;
        atomicAdd(&cnt[s], 1.0f);
    }
}

// ---------------------------------------------------------------------------
// Kernel 3: t = e + silu(concat(e, x[src], x[dst]) @ W1 + b1)  via bf16 WMMA
// Block: 64 edges x 256 dims, 256 threads = 8 waves in 4(M) x 2(N) layout.
// Each wave: 16x128 output tile = 8 accumulator fragments over 24 K-chunks.
// Software pipeline per K-chunk:
//   convert raw A (loaded last iter) -> bf16 fragment,
//   clause-load all 8 B fragments (dedicated regs),
//   issue next-chunk raw A loads,
//   drain the 8-WMMA chain (loads for kc+1 covered by the WMMA window).
// ---------------------------------------------------------------------------
__global__ __launch_bounds__(256, 2)
void k_gemm(const float* __restrict__ x, const float* __restrict__ e,
            const int* __restrict__ ei, const unsigned short* __restrict__ wBf,
            const float* __restrict__ b1, float* __restrict__ out) {
    const int lane = threadIdx.x & 31;
    const int wave = threadIdx.x >> 5;
    const int wm   = wave >> 1;          // 0..3  (M band of 16 edges)
    const int wn   = wave & 1;           // 0..1  (N band of 128 dims)
    const int h    = lane >> 4;          // lane half
    const int ml   = lane & 15;
    const int eBase = blockIdx.x * 64;
    const int edgeA = eBase + wm * 16 + ml;      // A-matrix row for this lane

    const int srcI = ei[edgeA];
    const int dstI = ei[E_EDGES + edgeA];
    const float* rowE = e + (size_t)edgeA * D_DIM;
    const float* rowS = x + (size_t)srcI * D_DIM;
    const float* rowD = x + (size_t)dstI * D_DIM;

    v8f zero = {};
    v8f acc[8];
#pragma unroll
    for (int i = 0; i < 8; ++i) acc[i] = zero;

    // prologue: raw A for kc = 0 (region 0 = e row, cb = 0)
    // per-lane A data: K = cb + {8h..8h+7} and cb + {16+8h..23+8h}
    float4 ar0 = *(const float4*)(rowE + 8 * h);
    float4 ar1 = *(const float4*)(rowE + 8 * h + 4);
    float4 ar2 = *(const float4*)(rowE + 16 + 8 * h);
    float4 ar3 = *(const float4*)(rowE + 16 + 8 * h + 4);

#pragma unroll
    for (int kc = 0; kc < 24; ++kc) {
        // 1) convert current raw A -> bf16 fragment (v_cvt_pk_bf16_f32 x8)
        v16bf a;
        a[0]  = (__bf16)ar0.x;  a[1]  = (__bf16)ar0.y;
        a[2]  = (__bf16)ar0.z;  a[3]  = (__bf16)ar0.w;
        a[4]  = (__bf16)ar1.x;  a[5]  = (__bf16)ar1.y;
        a[6]  = (__bf16)ar1.z;  a[7]  = (__bf16)ar1.w;
        a[8]  = (__bf16)ar2.x;  a[9]  = (__bf16)ar2.y;
        a[10] = (__bf16)ar2.z;  a[11] = (__bf16)ar2.w;
        a[12] = (__bf16)ar3.x;  a[13] = (__bf16)ar3.y;
        a[14] = (__bf16)ar3.z;  a[15] = (__bf16)ar3.w;

        // 2) load all 8 B fragments for this K-chunk into dedicated registers
        v16bf bf[8];
#pragma unroll
        for (int ns = 0; ns < 8; ++ns) {
            const unsigned short* bp =
                wBf + (((size_t)kc * 16 + wn * 8 + ns) * 32 + lane) * 16;
            bf[ns] = *(const v16bf*)bp;          // 32B contiguous, L2-resident
        }

        // 3) issue next chunk's raw A loads; waited on only at next convert
        if (kc < 23) {
            const int knext  = kc + 1;
            const int region = knext >> 3;        // 0:e  1:x[src]  2:x[dst]
            const float* rp  = (region == 0) ? rowE : ((region == 1) ? rowS : rowD);
            const int cb     = (knext & 7) * 32;
            ar0 = *(const float4*)(rp + cb + 8 * h);
            ar1 = *(const float4*)(rp + cb + 8 * h + 4);
            ar2 = *(const float4*)(rp + cb + 16 + 8 * h);
            ar3 = *(const float4*)(rp + cb + 16 + 8 * h + 4);
        }

        // 4) WMMA chain
#pragma unroll
        for (int ns = 0; ns < 8; ++ns) {
            acc[ns] = __builtin_amdgcn_wmma_f32_16x16x32_bf16(
                false, a, false, bf[ns], (short)0, acc[ns], false, false);
        }
    }

    // Epilogue: +b1, SiLU, +e residual; C layout: lane -> N=ml, reg r -> M=r+8h
#pragma unroll
    for (int ns = 0; ns < 8; ++ns) {
        const int dim = wn * 128 + ns * 16 + ml;
        const float bb = b1[dim];
#pragma unroll
        for (int r = 0; r < 8; ++r) {
            const int edge = eBase + wm * 16 + r + 8 * h;
            const float v = acc[ns][r] + bb;
            const float s = v / (1.0f + __expf(-v));     // SiLU
            const size_t o = (size_t)edge * D_DIM + dim;
            out[o] = e[o] + s;
        }
    }
}

// ---------------------------------------------------------------------------
// Kernel 4: segment sum & sum-of-squares of t.
// Block = 64 threads; thread owns one dim column of the LDS accumulators,
// so LDS read-modify-writes are race-free without atomics.
// grid = (64 edge slices, 4 dim chunks)
// ---------------------------------------------------------------------------
__global__ void k_stats(const float* __restrict__ t, const int* __restrict__ seg,
                        float* __restrict__ gsum, float* __restrict__ gsq) {
    __shared__ float sS[NGRAPH * 64];
    __shared__ float sQ[NGRAPH * 64];
    const int tid    = threadIdx.x;   // 0..63 = dim within chunk
    const int slice  = blockIdx.x;    // 0..63
    const int dchunk = blockIdx.y;    // 0..3
    for (int i = tid; i < NGRAPH * 64; i += 64) { sS[i] = 0.f; sQ[i] = 0.f; }
    __syncthreads();
    const int e0 = slice * (E_EDGES / 64);
    for (int k = 0; k < E_EDGES / 64; ++k) {
        const int edge = e0 + k;
        const int s = seg[edge];
        const float v = t[(size_t)edge * D_DIM + dchunk * 64 + tid];
        sS[s * 64 + tid] += v;
        sQ[s * 64 + tid] += v * v;
    }
    __syncthreads();
    for (int g = 0; g < NGRAPH; ++g) {
        atomicAdd(&gsum[g * D_DIM + dchunk * 64 + tid], sS[g * 64 + tid]);
        atomicAdd(&gsq [g * D_DIM + dchunk * 64 + tid], sQ[g * 64 + tid]);
    }
}

// ---------------------------------------------------------------------------
// Kernel 5: per-(graph,dim) affine params.
// out = w*(t - ms*mean)/std + b = t*scale + shift
// var = E[t^2] + mean^2*(ms^2 - 2ms)
// ---------------------------------------------------------------------------
__global__ void k_final(const float* __restrict__ cnt, const float* __restrict__ gsum,
                        const float* __restrict__ gsq, const float* __restrict__ wgt,
                        const float* __restrict__ bias, const float* __restrict__ msc,
                        float* __restrict__ scale, float* __restrict__ shift) {
    int i = blockIdx.x * 256 + threadIdx.x;   // 16384 exact
    int g = i >> 8, d = i & 255;
    float c    = fmaxf(cnt[g], 1.0f);
    float mean = gsum[i] / c;
    float ms   = msc[d];
    float var  = gsq[i] / c + mean * mean * (ms * ms - 2.0f * ms);
    float istd = rsqrtf(var + GN_EPS);
    scale[i] = wgt[d] * istd;
    shift[i] = bias[d] - wgt[d] * ms * mean * istd;
}

// ---------------------------------------------------------------------------
// Kernel 6: normalize in place
// ---------------------------------------------------------------------------
__global__ void k_apply(float* __restrict__ out, const int* __restrict__ seg,
                        const float* __restrict__ scale, const float* __restrict__ shift) {
    size_t i = (size_t)blockIdx.x * 256 + threadIdx.x;  // 160000*256 exact
    int edge = (int)(i >> 8);
    int d    = (int)(i & 255);
    int s    = seg[edge];
    out[i] = out[i] * scale[s * 256 + d] + shift[s * 256 + d];
}

extern "C" void kernel_launch(void* const* d_in, const int* in_sizes, int n_in,
                              void* d_out, int out_size, void* d_ws, size_t ws_size,
                              hipStream_t stream) {
    const float* x     = (const float*)d_in[0];
    const float* e     = (const float*)d_in[1];
    const int*   batch = (const int*)d_in[2];
    const int*   ei    = (const int*)d_in[3];
    const float* W1    = (const float*)d_in[4];
    const float* b1    = (const float*)d_in[5];
    const float* gw    = (const float*)d_in[6];
    const float* gb    = (const float*)d_in[7];
    const float* gm    = (const float*)d_in[8];
    float* out = (float*)d_out;
    char*  ws  = (char*)d_ws;

    unsigned short* wBf   = (unsigned short*)(ws + WS_WB);
    int*            seg   = (int*)(ws + WS_SEG);
    float*          cnt   = (float*)(ws + WS_CNT);
    float*          gsum  = (float*)(ws + WS_SUM);
    float*          gsq   = (float*)(ws + WS_SQ);
    float*          scale = (float*)(ws + WS_SCALE);
    float*          shift = (float*)(ws + WS_SHIFT);

    // zero the accumulated stats (cnt + sum + sumsq) each launch
    hipMemsetAsync(ws + WS_CNT, 0, 256 + 2 * NGRAPH * D_DIM * sizeof(float), stream);

    k_w1_to_frag<<<K_TOT * D_DIM / 256, 256, 0, stream>>>(W1, wBf);
    k_seg<<<(E_EDGES + 255) / 256, 256, 0, stream>>>(ei, batch, seg, cnt);
    k_gemm<<<E_EDGES / 64, 256, 0, stream>>>(x, e, ei, wBf, b1, out);
    k_stats<<<dim3(64, 4), 64, 0, stream>>>(out, seg, gsum, gsq);
    k_final<<<NGRAPH * D_DIM / 256, 256, 0, stream>>>(cnt, gsum, gsq, gw, gb, gm, scale, shift);
    k_apply<<<E_EDGES, 256, 0, stream>>>(out, seg, scale, shift);
}
